// EncoderGCN_70136815943923
// MI455X (gfx1250) — matrix operand order repaired and checked by reading, not compiled, not run on hardware
//
#include <hip/hip_runtime.h>

// ---------- types ----------
typedef __attribute__((ext_vector_type(16))) __bf16 v16bf;
typedef __attribute__((ext_vector_type(8)))  float  v8f;
typedef __attribute__((ext_vector_type(4)))  unsigned int u32x4;
typedef unsigned short u16;

// f32 -> bf16 round-to-nearest-even
static __device__ __forceinline__ u16 f2bf(float f) {
    union { float f; unsigned u; } v; v.f = f;
    unsigned x = v.u;
    unsigned r = x + 0x7FFFu + ((x >> 16) & 1u);
    return (u16)(r >> 16);
}

// ---------- elementwise kernels ----------
__global__ void k_cvt_bf16(const float* __restrict__ in, u16* __restrict__ out, long long n) {
    long long i = (long long)blockIdx.x * blockDim.x + threadIdx.x;
    if (i < n) out[i] = f2bf(in[i]);
}

__global__ void k_relu_cvt_bf16(const float* __restrict__ in, u16* __restrict__ out, long long n) {
    long long i = (long long)blockIdx.x * blockDim.x + threadIdx.x;
    if (i < n) out[i] = f2bf(fmaxf(in[i], 0.0f));
}

__global__ void k_fill(float* __restrict__ p, float v, long long n) {
    long long i = (long long)blockIdx.x * blockDim.x + threadIdx.x;
    if (i < n) p[i] = v;
}

// degree accumulation (self-loop already baked in as the 1.0 fill)
__global__ void k_deg(const int* __restrict__ dst, float* __restrict__ deg, long long E) {
    long long e = (long long)blockIdx.x * blockDim.x + threadIdx.x;
    if (e < E) unsafeAtomicAdd(&deg[dst[e]], 1.0f);
}

__global__ void k_dinv(float* __restrict__ d, long long n) {
    long long i = (long long)blockIdx.x * blockDim.x + threadIdx.x;
    if (i < n) d[i] = rsqrtf(fmaxf(d[i], 1.0f));
}

// agg[i,f] = y[i,f]*dinv[i]^2 + bias[f]   (self-loop term + bias)
__global__ void k_self_bias(const float* __restrict__ y, const float* __restrict__ dinv,
                            const float* __restrict__ bias, float* __restrict__ agg,
                            long long n, int fshift) {
    long long t = (long long)blockIdx.x * blockDim.x + threadIdx.x;
    if (t >= n) return;
    long long i = t >> fshift;
    int f = (int)(t & ((1 << fshift) - 1));
    float di = dinv[i];
    agg[t] = y[t] * di * di + bias[f];
}

// one wave per edge; lanes cover the feature dim; L2-resident f32 atomics
__global__ void k_edge_scatter(const int* __restrict__ src, const int* __restrict__ dst,
                               const float* __restrict__ dinv, const float* __restrict__ y,
                               float* __restrict__ agg, long long E, int fshift) {
    int lane = threadIdx.x & 31;
    long long wid = (((long long)blockIdx.x * blockDim.x) + threadIdx.x) >> 5;
    if (wid >= E) return;
    long long s = src[wid];
    long long d = dst[wid];
    float w = dinv[s] * dinv[d];
    int F = 1 << fshift;
    const float* ys = y + (s << fshift);
    float* ad = agg + (d << fshift);
    for (int f = lane; f < F; f += 32)
        unsafeAtomicAdd(&ad[f], ys[f] * w);
}

// ---------- WMMA GEMM: C(MxN) = A(MxK,bf16,row-major) * B(KxN,bf16,row-major) ----------
// One wave (block of 32) per 16-row strip x full N. NT = N/16 (compile-time).
// A is loaded exactly once; B fragments stay L1/L2-resident.
// Requires M%16==0, K%32==0, N==NT*16.
template <int NT>
__global__ void k_gemm_bf16_wmma(const u16* __restrict__ A, const u16* __restrict__ B,
                                 float* __restrict__ C, int M, int K) {
    const int N    = NT * 16;
    const int m0   = blockIdx.x * 16;
    const int lane = threadIdx.x;            // 0..31, full wave
    const int row  = lane & 15;              // A row within strip
    const int kb   = (lane >> 4) * 8;        // K sub-chunk select (ISA 16-bit A layout)

    union AFrag { v16bf v; u32x4 q[2]; };

    v8f acc[NT];
    #pragma unroll
    for (int t = 0; t < NT; ++t) acc[t] = (v8f){};

    const u16* Arow = A + (long long)(m0 + row) * K;
    for (int k0 = 0; k0 < K; k0 += 32) {
        AFrag a;
        // elems 0..7  = A[row, k0+kb .. k0+kb+7]       (VGPR0-3)
        // elems 8..15 = A[row, k0+16+kb .. k0+16+kb+7] (VGPR4-7)
        a.q[0] = *(const u32x4*)(Arow + k0 + kb);
        a.q[1] = *(const u32x4*)(Arow + k0 + 16 + kb);
        // B: lane holds column n=lane&15; elems 0..7 = k0+kb+s, elems 8..15 = k0+16+kb+(s-8)
        const u16* Bk = B + (long long)(k0 + kb) * N + (lane & 15);
        #pragma unroll
        for (int t = 0; t < NT; ++t) {
            union { v16bf v; u16 e[16]; } bu;
            #pragma unroll
            for (int s = 0; s < 16; ++s)
                bu.e[s] = Bk[(long long)(s < 8 ? s : s + 8) * N + t * 16];
            v16bf b = bu.v;
            acc[t] = __builtin_amdgcn_wmma_f32_16x16x32_bf16(
                /*neg_a=*/false, a.v, /*neg_b=*/false, b,
                /*c_mod=*/(short)0, acc[t], /*reuse_a=*/false, /*reuse_b=*/false);
        }
    }

    // D layout: lane 0-15 -> M=vgpr,  N=lane ; lane 16-31 -> M=vgpr+8, N=lane-16
    const int n  = lane & 15;
    const int mo = (lane >> 4) * 8;
    #pragma unroll
    for (int t = 0; t < NT; ++t) {
        float* Crow = C + (long long)(m0 + mo) * N + t * 16 + n;
        #pragma unroll
        for (int r = 0; r < 8; ++r)
            Crow[(long long)r * N] = acc[t][r];
    }
}

// ---------- host side ----------
static inline unsigned cdiv(long long a, long long b) { return (unsigned)((a + b - 1) / b); }

extern "C" void kernel_launch(void* const* d_in, const int* in_sizes, int n_in,
                              void* d_out, int out_size, void* d_ws, size_t ws_size,
                              hipStream_t stream) {
    const float* x  = (const float*)d_in[0];
    const int*   ei = (const int*)  d_in[1];   // [2,E] int32 (JAX default)
    const float* W1 = (const float*)d_in[2];
    const float* b1 = (const float*)d_in[3];
    const float* W2 = (const float*)d_in[4];
    const float* b2 = (const float*)d_in[5];

    const int       H   = in_sizes[3];                       // 128
    const int       Fo  = in_sizes[5];                       // 64
    const long long E   = (long long)in_sizes[1] / 2;        // 1.6M
    const int       Fin = in_sizes[2] / H;                   // 512
    const long long N   = (long long)in_sizes[0] / Fin;      // 100000

    int hs = 0; while ((1 << hs) < H)  ++hs;                 // 7
    int fs = 0; while ((1 << fs) < Fo) ++fs;                 // 6

    const int* src = ei;
    const int* dst = ei + E;

    // carve workspace (256B aligned)
    char* w = (char*)d_ws;
    auto carve = [&](size_t bytes) { char* p = w; w += (bytes + 255) & ~(size_t)255; return p; };
    float* dinv = (float*)carve(N * 4);
    u16*   xb   = (u16*)  carve((size_t)N * Fin * 2);
    u16*   W1b  = (u16*)  carve((size_t)Fin * H * 2);
    u16*   W2b  = (u16*)  carve((size_t)H * Fo * 2);
    float* y1   = (float*)carve((size_t)N * H * 4);
    float* agg1 = (float*)carve((size_t)N * H * 4);
    u16*   hb   = (u16*)  carve((size_t)N * H * 2);
    float* y2   = (float*)carve((size_t)N * Fo * 4);
    float* out  = (float*)d_out;

    const int T = 256;

    // 1) bf16 conversions
    k_cvt_bf16<<<cdiv(N * Fin, T), T, 0, stream>>>(x,  xb,  N * Fin);
    k_cvt_bf16<<<cdiv((long long)Fin * H, T), T, 0, stream>>>(W1, W1b, (long long)Fin * H);
    k_cvt_bf16<<<cdiv((long long)H * Fo, T), T, 0, stream>>>(W2, W2b, (long long)H * Fo);

    // 2) degrees -> dinv (shared by both layers)
    k_fill<<<cdiv(N, T), T, 0, stream>>>(dinv, 1.0f, N);
    k_deg <<<cdiv(E, T), T, 0, stream>>>(dst, dinv, E);
    k_dinv<<<cdiv(N, T), T, 0, stream>>>(dinv, N);

    // 3) layer 1: y1 = x @ W1  (WMMA, 16-row strip per wave, full N=128)
    k_gemm_bf16_wmma<8><<<(unsigned)(N / 16), 32, 0, stream>>>(xb, W1b, y1, (int)N, Fin);
    k_self_bias<<<cdiv(N * H, T), T, 0, stream>>>(y1, dinv, b1, agg1, N * H, hs);
    k_edge_scatter<<<cdiv(E * 32, T), T, 0, stream>>>(src, dst, dinv, y1, agg1, E, hs);
    k_relu_cvt_bf16<<<cdiv(N * H, T), T, 0, stream>>>(agg1, hb, N * H);

    // 4) layer 2: y2 = h @ W2 (WMMA, full N=64), self-loop+bias into d_out, edge scatter
    k_gemm_bf16_wmma<4><<<(unsigned)(N / 16), 32, 0, stream>>>(hb, W2b, y2, (int)N, H);
    k_self_bias<<<cdiv(N * Fo, T), T, 0, stream>>>(y2, dinv, b2, out, N * Fo, fs);
    k_edge_scatter<<<cdiv(E * 32, T), T, 0, stream>>>(src, dst, dinv, y2, out, E, fs);
}
